// OutputPPBlock_3023656976808
// MI455X (gfx1250) — matrix operand order repaired and checked by reading, not compiled, not run on hardware
//
#include <hip/hip_runtime.h>
#include <hip/hip_bf16.h>
#include <math.h>

// ---------------- problem constants (fixed by the reference) ----------------
#define EDGES      800000
#define NNODES     50000
#define NUM_RADIAL 6
#define HIDDEN     256
#define OUT_EMB    192
#define NLAYERS    3

typedef __attribute__((ext_vector_type(16))) _Float16 v16h;
typedef __attribute__((ext_vector_type(8)))  _Float16 v8h;
typedef __attribute__((ext_vector_type(4)))  _Float16 v4h;
typedef __attribute__((ext_vector_type(8)))  float    v8f;
typedef __attribute__((ext_vector_type(4)))  float    v4f;

// ---------------------------------------------------------------------------
// WMMA helper: D = A(16x32 f16) * B(32x16 f16) + C(16x16 f32)
// ---------------------------------------------------------------------------
__device__ __forceinline__ v8f wmma_f16(v16h a, v16h b, v8f c) {
  return __builtin_amdgcn_wmma_f32_16x16x32_f16(
      /*neg_a=*/false, a, /*neg_b=*/false, b,
      /*c_mod=*/(short)0, c, /*reuse_a=*/false, /*reuse_b=*/false);
}

// A-fragment gather from LDS, row-major f16 matrix.
// Lane l holds row m=l&15; lo lanes cover K = kc*32 + 0..15,
// hi lanes K = kc*32 + 16..31 (matches B fragment packing).
// Both halves are contiguous 16B runs -> two ds_load_b128 per fragment.
__device__ __forceinline__ v16h load_a_frag(const _Float16* row, int kc, bool hi) {
  const _Float16* p = row + kc * 32 + (hi ? 16 : 0);
  v8h lo = *(const v8h*)p;
  v8h up = *(const v8h*)(p + 8);
  return __builtin_shufflevector(lo, up, 0,1,2,3,4,5,6,7,8,9,10,11,12,13,14,15);
}

// ---------------------------------------------------------------------------
// One GEMM stage: [16 x (KC*32)] @ [(KC*32) x 192] + bias (+SiLU), f16 in LDS.
// Split into two passes of 6 N-tiles so only 48 accumulator VGPRs are live,
// leaving registers free for the scheduler to keep ~6 B fragments in flight.
// rd and wr must be DIFFERENT LDS buffers (ping-pong), so pass 1's stores
// don't conflict with pass 2's A reads.
// ---------------------------------------------------------------------------
template <int KC, bool ACT>
__device__ __forceinline__ void gemm_stage(const _Float16* __restrict__ rd, int strideK,
                                           const v16h* __restrict__ wf,
                                           const float* __restrict__ bias,
                                           _Float16* __restrict__ wr, int lane) {
  const bool hi  = lane >= 16;
  const int  l15 = lane & 15;
  const _Float16* aRow = rd + l15 * strideK;
  #pragma unroll
  for (int half = 0; half < 2; ++half) {
    const v8f zero = {};
    v8f acc[6];
    #pragma unroll
    for (int t = 0; t < 6; ++t) acc[t] = zero;
    #pragma unroll
    for (int kc = 0; kc < KC; ++kc) {
      v16h a = load_a_frag(aRow, kc, hi);
      if (kc + 1 < KC)
        __builtin_prefetch(&wf[((kc + 1) * 12 + half * 6) * 32 + lane], 0, 0);
      #pragma unroll
      for (int nt = 0; nt < 6; ++nt) {
        v16h b = wf[(kc * 12 + half * 6 + nt) * 32 + lane];
        acc[nt] = wmma_f16(a, b, acc[nt]);
      }
    }
    // epilogue: bias (+SiLU), write f16 to the OTHER LDS buffer.
    // C/D layout: element (m,n) with m = v + (hi?8:0), n = tile*16 + (l&15).
    #pragma unroll
    for (int nt = 0; nt < 6; ++nt) {
      int n = (half * 6 + nt) * 16 + l15;
      float bb = bias[n];
      #pragma unroll
      for (int v = 0; v < 8; ++v) {
        float hv = acc[nt][v] + bb;
        if (ACT) hv = hv * (1.f / (1.f + __expf(-hv)));   // SiLU
        wr[(v + (hi ? 8 : 0)) * OUT_EMB + n] = (_Float16)hv;
      }
    }
  }
}

// ---------------------------------------------------------------------------
// Kernel 0: zero the node accumulator (xn) — non-temporal stores (pure stream)
// ---------------------------------------------------------------------------
__global__ __launch_bounds__(256) void zero_f4_kernel(v4f* p, int n4) {
  int stride = gridDim.x * blockDim.x;
  const v4f z = {};
  for (int i = blockIdx.x * blockDim.x + threadIdx.x; i < n4; i += stride)
    __builtin_nontemporal_store(z, p + i);
}

// ---------------------------------------------------------------------------
// Kernel 1: convert a row-major f32 weight [K x Nn] into f16 *fragment-major*
// B-operand layout: fragment (kc,nt) = 512 halves, lane l owns halves
// [l*16 .. l*16+15] with n = nt*16 + (l&15), k = kc*32 + (l>=16?16:0) + h.
// ---------------------------------------------------------------------------
__global__ __launch_bounds__(256) void convert_wfrag_kernel(const float* __restrict__ src,
                                                            _Float16* __restrict__ dst,
                                                            int K, int Nn) {
  int NT = Nn >> 4;
  int total = K * Nn;
  int stride = gridDim.x * blockDim.x;
  for (int t = blockIdx.x * blockDim.x + threadIdx.x; t < total; t += stride) {
    int frag = t >> 9, r = t & 511;
    int l = r >> 4, h = r & 15;
    int kc = frag / NT, nt = frag % NT;
    int k = kc * 32 + ((l >= 16) ? 16 : 0) + h;
    int n = nt * 16 + (l & 15);
    dst[t] = (_Float16)src[k * Nn + n];
  }
}

// ---------------------------------------------------------------------------
// Kernel 2: edge phase.  One wave per edge:
//   g = rbf[e] @ W_rbf  (lane's 6x8 W_rbf slab hoisted into 48 registers)
//   xe = g * x[e]       (non-temporal: x is 819 MB touched exactly once)
//   xn[i[e]] += xe      (hardware f32 atomics; xn stays hot in L2)
// ---------------------------------------------------------------------------
__global__ __launch_bounds__(256) void edge_kernel(const float* __restrict__ x,
                                                   const float* __restrict__ rbf,
                                                   const int*   __restrict__ idx,
                                                   const float* __restrict__ W_rbf,
                                                   float* __restrict__ xn, int E) {
  const int lane = threadIdx.x & 31;
  const int wave = (int)((blockIdx.x * blockDim.x + threadIdx.x) >> 5);
  const int nwaves = (int)((gridDim.x * blockDim.x) >> 5);
  const int c0 = lane * 8;                       // 8 hidden columns per lane

  float w[NUM_RADIAL][8];
  #pragma unroll
  for (int r = 0; r < NUM_RADIAL; ++r)
    #pragma unroll
    for (int j = 0; j < 8; ++j)
      w[r][j] = W_rbf[r * HIDDEN + c0 + j];

  for (int e = wave; e < E; e += nwaves) {
    float rb[NUM_RADIAL];
    #pragma unroll
    for (int r = 0; r < NUM_RADIAL; ++r)
      rb[r] = __builtin_nontemporal_load(rbf + (size_t)e * NUM_RADIAL + r);
    const int node = idx[e];

    const v4f* xp = (const v4f*)(x + (size_t)e * HIDDEN + c0);
    v4f x0 = __builtin_nontemporal_load(xp);
    v4f x1 = __builtin_nontemporal_load(xp + 1);
    float xv[8] = {x0[0], x0[1], x0[2], x0[3], x1[0], x1[1], x1[2], x1[3]};

    float* dst = xn + (size_t)node * HIDDEN + c0;
    #pragma unroll
    for (int j = 0; j < 8; ++j) {
      float g = 0.f;
      #pragma unroll
      for (int r = 0; r < NUM_RADIAL; ++r) g = fmaf(rb[r], w[r][j], g);
      atomicAdd(dst + j, g * xv[j]);
    }
  }
}

// ---------------------------------------------------------------------------
// Kernel 3: fused node MLP, one wave (32 threads) per 16-node tile.
// Activations ping-pong between two LDS buffers between GEMM stages.
// ---------------------------------------------------------------------------
__global__ __launch_bounds__(32) void node_mlp_kernel(const float*    __restrict__ xn,
                                                      const _Float16* __restrict__ wup,
                                                      const _Float16* __restrict__ wlin,
                                                      const float*    __restrict__ b_up,
                                                      const float*    __restrict__ b_lins,
                                                      const float*    __restrict__ W_out,
                                                      float* __restrict__ out) {
  __shared__ __align__(16) _Float16 smA[16 * HIDDEN];   // 8 KB: xn stage, then ping
  __shared__ __align__(16) _Float16 smB[16 * OUT_EMB];  // 6 KB: pong

  const int lane     = threadIdx.x;
  const int nodeBase = blockIdx.x * 16;                 // N = 3125 * 16 exactly

  // ---- stage xn tile: 16x256 f32 -> f16 row-major (stride 256) in smA ----
  const float* src = xn + (size_t)nodeBase * HIDDEN;
  #pragma unroll
  for (int j = 0; j < 32; ++j) {
    int f4 = j * 32 + lane;                             // 1024 float4 total
    v4f v = *(const v4f*)(src + f4 * 4);
    v4h hv = {(_Float16)v[0], (_Float16)v[1], (_Float16)v[2], (_Float16)v[3]};
    *(v4h*)(smA + f4 * 4) = hv;
  }
  __syncthreads();

  // ---- up-projection: smA (stride 256) -> smB, bias only ----
  gemm_stage<8, false>(smA, HIDDEN, (const v16h*)wup, b_up, smB, lane);
  __syncthreads();

  // ---- 3 hidden layers: ping-pong smB -> smA -> smB -> smA ----
  const _Float16* rbuf = smB;
  _Float16*       wbuf = smA;
  for (int layer = 0; layer < NLAYERS; ++layer) {
    const v16h* wf = (const v16h*)(wlin + (size_t)layer * OUT_EMB * OUT_EMB);
    gemm_stage<6, true>(rbuf, OUT_EMB, wf, b_lins + layer * OUT_EMB, wbuf, lane);
    __syncthreads();
    const _Float16* t = rbuf; rbuf = wbuf; wbuf = (_Float16*)t;
  }
  // after 3 swaps, final activations are in smA (== rbuf)

  // ---- output head: [16x192] @ [192x1] ----
  if (lane < 16) {
    const _Float16* hr = rbuf + lane * OUT_EMB;
    float s = 0.f;
    #pragma unroll 4
    for (int n = 0; n < OUT_EMB; ++n) s = fmaf((float)hr[n], W_out[n], s);
    out[nodeBase + lane] = s;
  }
}

// ---------------------------------------------------------------------------
// Host-side launcher (graph-capture safe: only kernel launches on `stream`)
// ---------------------------------------------------------------------------
extern "C" void kernel_launch(void* const* d_in, const int* in_sizes, int n_in,
                              void* d_out, int out_size, void* d_ws, size_t ws_size,
                              hipStream_t stream) {
  const float* x      = (const float*)d_in[0];   // [E, 256]
  const float* rbf    = (const float*)d_in[1];   // [E, 6]
  const int*   idx    = (const int*)  d_in[2];   // [E]
  /* d_in[3] = num_nodes (compile-time constant NNODES) */
  const float* W_rbf  = (const float*)d_in[4];   // [6, 256]
  const float* W_up   = (const float*)d_in[5];   // [256, 192]
  const float* b_up   = (const float*)d_in[6];   // [192]
  const float* W_lins = (const float*)d_in[7];   // [3, 192, 192]
  const float* b_lins = (const float*)d_in[8];   // [3, 192]
  const float* W_out  = (const float*)d_in[9];   // [192, 1]
  float* out = (float*)d_out;                    // [N, 1]

  // workspace layout
  char* ws = (char*)d_ws;
  float*    xn     = (float*)ws;                                   // N*256 f32 (51.2 MB)
  _Float16* wup_h  = (_Float16*)(ws + (size_t)NNODES * HIDDEN * 4);// 256*192 f16
  _Float16* wlin_h = wup_h + (size_t)HIDDEN * OUT_EMB;             // 3*192*192 f16

  // 1) zero node accumulator (must happen every call: atomics accumulate)
  zero_f4_kernel<<<2048, 256, 0, stream>>>((v4f*)xn, NNODES * HIDDEN / 4);

  // 2) pre-convert weights to f16 fragment-major (B-operand layout)
  convert_wfrag_kernel<<<192, 256, 0, stream>>>(W_up, wup_h, HIDDEN, OUT_EMB);
  for (int l = 0; l < NLAYERS; ++l)
    convert_wfrag_kernel<<<144, 256, 0, stream>>>(W_lins + (size_t)l * OUT_EMB * OUT_EMB,
                                                  wlin_h + (size_t)l * OUT_EMB * OUT_EMB,
                                                  OUT_EMB, OUT_EMB);

  // 3) edge phase: gate + scatter-add (memory/atomic bound, ~870 MB traffic)
  edge_kernel<<<4096, 256, 0, stream>>>(x, rbf, idx, W_rbf, xn, EDGES);

  // 4) node phase: fused WMMA MLP, one wave per 16-node tile (3125 tiles)
  node_mlp_kernel<<<NNODES / 16, 32, 0, stream>>>(xn, wup_h, wlin_h,
                                                  b_up, b_lins, W_out, out);
}
